// AnemllQATLinear_44152263803421
// MI455X (gfx1250) — compile-verified, matrix operand order checked by reading, not run on hardware
//
#include <hip/hip_runtime.h>

typedef __attribute__((ext_vector_type(16))) _Float16 v16h;
typedef __attribute__((ext_vector_type(8)))  _Float16 v8h;
typedef __attribute__((ext_vector_type(4)))  _Float16 v4h;
typedef __attribute__((ext_vector_type(8)))  float    v8f;
typedef __attribute__((ext_vector_type(4)))  float    v4f;

// b128 payload type exactly as the async-LDS builtin expects it
typedef int v4i_vs __attribute__((vector_size(16)));
typedef __attribute__((address_space(1))) v4i_vs* gb128_t;   // global
typedef __attribute__((address_space(3))) v4i_vs* lb128_t;   // LDS

#define O_DIM 4096
#define I_DIM 4096
#define GS_Q 128
#define G_DIM 32            // I_DIM / GS_Q
#define RANK 4
#define LUT_SIZE 16
#define LORA_R 16
#define LORA_SCALING 2.0f   // 32 / 16
#define EPS_Q 1e-8f

// --------------------- CDNA5 async global->LDS support ---------------------
#if defined(__gfx1250__) && __has_builtin(__builtin_amdgcn_global_load_async_to_lds_b128)
#define HAVE_ASYNC_LDS 1
#else
#define HAVE_ASYNC_LDS 0
#endif

template <int N>
__device__ __forceinline__ void wait_asynccnt() {
#if HAVE_ASYNC_LDS
#if __has_builtin(__builtin_amdgcn_s_wait_asynccnt)
    __builtin_amdgcn_s_wait_asynccnt(N);
#else
    asm volatile("s_wait_asynccnt %0" ::"i"(N));
#endif
#endif
}

// ---------------------------------------------------------------------------
// Kernel A: w_eff[o][i] = dequant(weight) + LORA_SCALING * (lora_B @ lora_A)
// folded once into an f16 weight matrix (row-major [O][I], K contiguous).
// ---------------------------------------------------------------------------
__global__ __launch_bounds__(256) void build_weff_f16(
    const float* __restrict__ weight,
    const float* __restrict__ scale_A,   // [O, RANK]
    const float* __restrict__ scale_B,   // [RANK, G]
    const float* __restrict__ lut,       // [16]
    const float* __restrict__ lora_A,    // [LORA_R, I]
    const float* __restrict__ lora_B,    // [O, LORA_R]
    _Float16* __restrict__ weff)         // [O, I]
{
    int t  = blockIdx.x * blockDim.x + threadIdx.x;  // one thread per 4 elems
    int o  = t / (I_DIM / 4);
    int i4 = (t % (I_DIM / 4)) * 4;
    int g  = i4 / GS_Q;                  // 4 consecutive i share one group

    float s = 0.0f;
#pragma unroll
    for (int r = 0; r < RANK; ++r)
        s += scale_A[o * RANK + r] * scale_B[r * G_DIM + g];
    s = fmaxf(s, EPS_Q);
    const float inv_s    = 1.0f / s;
    const float inv_step = (LUT_SIZE - 1) * 0.5f;    // 1/(2/(L-1))

    float lb[LORA_R];
#pragma unroll
    for (int r = 0; r < LORA_R; ++r) lb[r] = lora_B[o * LORA_R + r];

    v4h out;
#pragma unroll
    for (int j = 0; j < 4; ++j) {
        const int i = i4 + j;
        float w   = weight[(size_t)o * I_DIM + i];
        float nrm = fminf(fmaxf(w * inv_s, -1.0f), 1.0f);
        int   idx = (int)rintf((nrm + 1.0f) * inv_step);   // round-to-even
        idx = min(max(idx, 0), LUT_SIZE - 1);
        float dq  = lut[idx] * s;
        float ld  = 0.0f;
#pragma unroll
        for (int r = 0; r < LORA_R; ++r)
            ld = fmaf(lb[r], lora_A[(size_t)r * I_DIM + i], ld);
        out[j] = (_Float16)(dq + LORA_SCALING * ld);
    }
    *(v4h*)(weff + (size_t)o * I_DIM + i4) = out;   // 8B aligned store
}

// ---------------------------------------------------------------------------
// Kernel B: y[M,N] = x[M,K] @ weff[N,K]^T + bias[N]
// 8 waves/block (wave32), block tile 128x256, BK=32, double-buffered LDS.
// B tiles: GLOBAL_LOAD_ASYNC_TO_LDS_B128 (ASYNCcnt).  A tiles: register-
// pipelined f32->f16 conversion.  Wave tile 64x64 via 4x4 WMMA accumulators.
// ---------------------------------------------------------------------------
#define BM 128
#define BN 256
#define BK 32
#define LPITCH (BK + 8)     // 40 halves = 80B row pitch (16B aligned)

__global__ __launch_bounds__(256, 1) void qat_gemm_f16(
    const float*    __restrict__ x,     // [M, K] f32
    const _Float16* __restrict__ wq,    // [N, K] f16
    const float*    __restrict__ bias,  // [N]
    float*          __restrict__ y,     // [M, N]
    int M, int N, int K)
{
    __shared__ _Float16 sA[2 * BM * LPITCH];   // 20,480 B
    __shared__ _Float16 sB[2 * BN * LPITCH];   // 40,960 B

    const int tid  = threadIdx.x;
    const int lane = tid & 31;
    const int wave = tid >> 5;        // 0..7
    const int wm   = wave & 1;        // 0..1 -> 64-row slab
    const int wn   = wave >> 1;       // 0..3 -> 64-col slab
    const int m0   = blockIdx.y * BM;
    const int n0   = blockIdx.x * BN;

    const int lm = lane & 15;         // row within 16x16 tile
    const int lh = lane >> 4;         // half-wave selector

    // staging maps: A 128x32 f32 (4 rounds, v4f/thread), B 256x32 f16 (4
    // rounds, 16B/thread == one async_to_lds_b128 per wave per round)
    const int a_row = tid >> 3;            // 0..31
    const int a_col = (tid & 7) << 2;      // 0..28
    const int b_row = tid >> 2;            // 0..63
    const int b_col = (tid & 3) << 3;      // 0..24

    v8f acc[4][4];
#pragma unroll
    for (int i = 0; i < 4; ++i)
#pragma unroll
        for (int j = 0; j < 4; ++j) acc[i][j] = (v8f){};

    v4f areg[4];
    auto loadA = [&](int k0) {
#pragma unroll
        for (int r = 0; r < 4; ++r)
            areg[r] = *(const v4f*)(x + (size_t)(m0 + r * 32 + a_row) * K
                                    + k0 + a_col);
    };
    auto storeA = [&](_Float16* buf) {
#pragma unroll
        for (int r = 0; r < 4; ++r) {
            v4h h = { (_Float16)areg[r][0], (_Float16)areg[r][1],
                      (_Float16)areg[r][2], (_Float16)areg[r][3] };
            *(v4h*)(buf + (r * 32 + a_row) * LPITCH + a_col) = h;
        }
    };
    auto stageB = [&](int k0, _Float16* buf) {
#pragma unroll
        for (int r = 0; r < 4; ++r) {
            const _Float16* g = wq + (size_t)(n0 + r * 64 + b_row) * K
                                + k0 + b_col;
            _Float16* l = buf + (r * 64 + b_row) * LPITCH + b_col;
#if HAVE_ASYNC_LDS
            __builtin_amdgcn_global_load_async_to_lds_b128(
                (gb128_t)g, (lb128_t)l, /*imm offset*/ 0, /*cpol*/ 0);
#else
            *(v8h*)l = *(const v8h*)g;
#endif
        }
    };

    const int NT = K / BK;            // 128 stages

    // ---- prologue: stage 0 in flight
    loadA(0);
    stageB(0, sB);

#pragma unroll 1
    for (int kt = 0; kt < NT; ++kt) {
        const int cur = kt & 1;
        _Float16* bufA = sA + cur * BM * LPITCH;
        _Float16* bufB = sB + cur * BN * LPITCH;

        storeA(bufA);                                  // stage kt A -> LDS
        if (kt + 1 < NT) {                             // issue stage kt+1
            loadA((kt + 1) * BK);
            stageB((kt + 1) * BK, sB + (cur ^ 1) * BN * LPITCH);
            wait_asynccnt<4>();    // retire stage-kt asyncs, keep 4 in flight
        } else {
            wait_asynccnt<0>();
        }
        __syncthreads();

        // ---- compute: one K=32 WMMA step over 16 accumulator tiles
        v16h af[4];
#pragma unroll
        for (int i = 0; i < 4; ++i) {
            // lane lm = M row; VGPR0-3 hold K = lh*8+0..7,
            // VGPR4-7 hold K = 16+lh*8+0..7  (two ds_load_b128)
            const _Float16* p = bufA + (wm * 64 + i * 16 + lm) * LPITCH
                                + lh * 8;
            v8h lo = *(const v8h*)p;
            v8h hi = *(const v8h*)(p + 16);
            v16h a;
#pragma unroll
            for (int e = 0; e < 8; ++e) { a[e] = lo[e]; a[e + 8] = hi[e]; }
            af[i] = a;
        }
#pragma unroll
        for (int j = 0; j < 4; ++j) {
            const _Float16* p = bufB + (wn * 64 + j * 16 + lm) * LPITCH
                                + lh * 8;
            v8h lo = *(const v8h*)p;
            v8h hi = *(const v8h*)(p + 16);
            v16h b;
#pragma unroll
            for (int e = 0; e < 8; ++e) { b[e] = lo[e]; b[e + 8] = hi[e]; }
#pragma unroll
            for (int i = 0; i < 4; ++i)
                acc[i][j] = __builtin_amdgcn_wmma_f32_16x16x32_f16(
                    false, af[i], false, b, (short)0, acc[i][j],
                    false, false);
        }
        __syncthreads();   // protect buffers before next stage overwrites
    }

    // ---- epilogue: C/D layout -> lane (lm,lh): VGPR e = row e + lh*8, col lm
#pragma unroll
    for (int i = 0; i < 4; ++i) {
        const int mbase = m0 + wm * 64 + i * 16 + lh * 8;
#pragma unroll
        for (int j = 0; j < 4; ++j) {
            const int col = n0 + wn * 64 + j * 16 + lm;
            const float bv = bias[col];
#pragma unroll
            for (int e = 0; e < 8; ++e)
                y[(size_t)(mbase + e) * N + col] = acc[i][j][e] + bv;
        }
    }
}

// ---------------------------------------------------------------------------
extern "C" void kernel_launch(void* const* d_in, const int* in_sizes, int n_in,
                              void* d_out, int out_size, void* d_ws,
                              size_t ws_size, hipStream_t stream)
{
    const float* x       = (const float*)d_in[0];
    const float* weight  = (const float*)d_in[1];
    const float* bias    = (const float*)d_in[2];
    const float* scale_A = (const float*)d_in[3];
    const float* scale_B = (const float*)d_in[4];
    const float* lut     = (const float*)d_in[5];
    const float* lora_A  = (const float*)d_in[6];
    const float* lora_B  = (const float*)d_in[7];
    float*       y       = (float*)d_out;
    _Float16*    weff    = (_Float16*)d_ws;   // 4096*4096*2 = 32 MB scratch

    const int M = in_sizes[0] / I_DIM;        // 4*4096 = 16384 tokens

    {   // fold quantization + LoRA into one f16 weight
        int total = O_DIM * I_DIM / 4;
        build_weff_f16<<<dim3(total / 256), dim3(256), 0, stream>>>(
            weight, scale_A, scale_B, lut, lora_A, lora_B, weff);
    }
    {   // single fused GEMM + bias
        dim3 grid(O_DIM / BN, M / BM);        // (16, 128)
        qat_gemm_f16<<<grid, dim3(256), 0, stream>>>(
            x, weff, bias, y, M, O_DIM, I_DIM);
    }
}
